// WindowSelfAttentionModule_45741401702826
// MI455X (gfx1250) — matrix operand order repaired and checked by reading, not compiled, not run on hardware
//
#include <hip/hip_runtime.h>
#include <hip/hip_bf16.h>
#include <cmath>

// ---------------------------------------------------------------------------
// Types for gfx1250 WMMA (wave32): A/B are 16 f16 per lane, C/D are 8 f32.
// ---------------------------------------------------------------------------
typedef _Float16 v16h __attribute__((ext_vector_type(16)));
typedef _Float16 v8h  __attribute__((ext_vector_type(8)));
typedef float    v8f  __attribute__((ext_vector_type(8)));

#define WMMA_F16(a, b, c) \
  __builtin_amdgcn_wmma_f32_16x16x32_f16(false, (a), false, (b), (short)0, (c), false, false)

#define LN100 4.6051702f

// A-matrix fragment (16-bit, 16x32 MxK). Lane L: M = m0 + (L&15), half = L>>4.
// Element i -> K = k0 + i + 8*((i>>3) + half): two contiguous 8-half runs.
__device__ __forceinline__ v16h frag_a(const _Float16* base, int stride, int m0, int k0, int lane) {
  const _Float16* p = base + (m0 + (lane & 15)) * stride + k0 + 8 * (lane >> 4);
  v8h lo = *(const v8h*)(p);        // K = 8*half .. 8*half+7
  v8h hi = *(const v8h*)(p + 16);   // K = 16+8*half .. 16+8*half+7
  v16h r;
#pragma unroll
  for (int i = 0; i < 8; ++i) { r[i] = lo[i]; r[i + 8] = hi[i]; }
  return r;
}

// B-matrix fragment (16-bit, 32x16 KxN) from an N-major ([n][k]) buffer.
// Lane L: N = n0 + (L&15), element i -> K = k0 + i + 8*((i>>3) + (L>>4)) (same as A).
__device__ __forceinline__ v16h frag_b(const _Float16* base, int stride, int n0, int k0, int lane) {
  const _Float16* p = base + (n0 + (lane & 15)) * stride + k0 + 8 * (lane >> 4);
  v8h lo = *(const v8h*)(p);
  v8h hi = *(const v8h*)(p + 16);
  v16h r;
#pragma unroll
  for (int i = 0; i < 8; ++i) { r[i] = lo[i]; r[i + 8] = hi[i]; }
  return r;
}

// ---------------------------------------------------------------------------
// Prep kernels (tiny): transpose+convert weights to f16, build CPB bias.
// ---------------------------------------------------------------------------
__global__ void prep_weights(const float* __restrict__ wqkv, const float* __restrict__ wout,
                             _Float16* __restrict__ wqkvT, _Float16* __restrict__ woutT) {
  const int total1 = 512 * 1536, total2 = 512 * 512;
  for (int i = blockIdx.x * blockDim.x + threadIdx.x; i < total1 + total2;
       i += gridDim.x * blockDim.x) {
    if (i < total1) {
      int n = i / 512, k = i % 512;                 // wqkvT[n][k] = wqkv[k][n]
      wqkvT[i] = (_Float16)wqkv[k * 1536 + n];
    } else {
      int j = i - total1;
      int n = j / 512, k = j % 512;                 // woutT[n][k] = wout[k][n]
      woutT[j] = (_Float16)wout[k * 512 + n];
    }
  }
}

__device__ __forceinline__ float cpb_coord(int a) {  // SwinV2 log-spaced coord
  float g = (float)(a - 7) * (8.0f / 7.0f);
  float s = (g >= 0.0f) ? 1.0f : -1.0f;
  return s * log2f(fabsf(g) + 1.0f) * (1.0f / 3.0f);  // /log2(8)
}

__global__ void cpb_table(const float* __restrict__ w1, const float* __restrict__ w2,
                          float* __restrict__ btab) {
  int r = blockIdx.x;            // 0..224 table rows
  int hh = threadIdx.x;          // 0..15 heads
  float t0 = cpb_coord(r / 15);
  float t1 = cpb_coord(r % 15);
  float acc = 0.0f;
  for (int j = 0; j < 512; ++j) {
    float hsum = t0 * w1[j] + t1 * w1[512 + j];
    float hs = hsum * fminf(fmaxf(hsum + 3.0f, 0.0f), 6.0f) * (1.0f / 6.0f);  // hardswish
    acc += hs * w2[j * 16 + hh];
  }
  btab[r * 16 + hh] = acc;
}

__global__ void bias_sigmoid(const float* __restrict__ btab, float* __restrict__ b16) {
  int e = blockIdx.x * blockDim.x + threadIdx.x;      // [h][n][m] flat, 65536
  if (e >= 16 * 64 * 64) return;
  int hh = e >> 12, n = (e >> 6) & 63, m = e & 63;
  int di = (n >> 3) - (m >> 3) + 7;
  int dj = (n & 7) - (m & 7) + 7;
  float bv = btab[(di * 15 + dj) * 16 + hh];
  b16[e] = 16.0f / (1.0f + __expf(-bv));
}

// ---------------------------------------------------------------------------
// Fused window attention. grid = 2048 (one window per block), 512 thr = 16 waves.
// 320 KB dynamic LDS (full WGP), regions overlapped across stages (halves):
//   [0,32768)       : Xf16            / attn lo    / AO
//   [32768,98304)   : QK [64][1024]   / attn hi + P lo
//   [98304,131072)  : (spare)         / P hi
//   [131072,163840) : VT [512][64]
// ---------------------------------------------------------------------------
__global__ __launch_bounds__(512)
void fused_window_attn(const float* __restrict__ x, const float* __restrict__ mask,
                       const _Float16* __restrict__ wqkvT, const _Float16* __restrict__ woutT,
                       const float* __restrict__ bias16, const float* __restrict__ lscale,
                       float* __restrict__ out) {
  extern __shared__ _Float16 smem[];
  _Float16* Xs  = smem;             // [64][512]
  _Float16* QKs = smem + 32768;     // [64][1024]: q cols 0..511, k cols 512..1023
  _Float16* VTs = smem + 131072;    // [512][64]  (transposed V)
  _Float16* Atn = smem;             // [64][64][16]  logits, head-minor
  _Float16* Ps  = smem + 65536;     // [16][64][64]  softmax probs
  _Float16* AOs = smem;             // [64][512]     attention output

  const int tid  = threadIdx.x;
  const int wave = tid >> 5;        // 0..15 == head
  const int lane = tid & 31;
  const int b    = blockIdx.x;
  const int w    = b & 63;          // window index for mask

  // ---- Stage 0: X tile -> f16 LDS ---------------------------------------
  {
    const float* xb = x + (size_t)b * 64 * 512;
    for (int i = tid; i < 64 * 512; i += 512) Xs[i] = (_Float16)xb[i];
  }
  __syncthreads();

  // ---- Stage 1: QKV = X @ Wqkv (M=64,K=512,N=1536), route to QK/VT ------
  // Wave handles 6 column tiles; 4 accumulators share each B fragment.
  for (int j = 0; j < 6; ++j) {
    const int ct = wave * 6 + j;                   // 96 col tiles / 16 waves
    v8f acc[4] = {};
#pragma unroll 2
    for (int kt = 0; kt < 16; ++kt) {
      v16h bb = frag_b(wqkvT, 512, ct * 16, kt * 32, lane);
#pragma unroll
      for (int mt = 0; mt < 4; ++mt) {
        v16h a = frag_a(Xs, 512, mt * 16, kt * 32, lane);
        acc[mt] = WMMA_F16(a, bb, acc[mt]);
      }
    }
    // ct is wave-uniform: force a scalar branch (whole 16-col tile routes
    // to one region), avoiding per-element exec-mask divergence.
    const int ctu = __builtin_amdgcn_readfirstlane(ct);
    if (ctu < 64) {                                 // Q or K -> QK buffer
      int c = ct * 16 + (lane & 15);                // 0..1023
#pragma unroll
      for (int mt = 0; mt < 4; ++mt)
#pragma unroll
        for (int r = 0; r < 8; ++r) {
          int n = mt * 16 + r + 8 * (lane >> 4);
          QKs[n * 1024 + c] = (_Float16)acc[mt][r];
        }
    } else {                                        // V -> transposed VT
      int cv = (ct - 64) * 16 + (lane & 15);        // 0..511
#pragma unroll
      for (int mt = 0; mt < 4; ++mt)
#pragma unroll
        for (int r = 0; r < 8; ++r) {
          int n = mt * 16 + r + 8 * (lane >> 4);
          VTs[cv * 64 + n] = (_Float16)acc[mt][r];
        }
    }
  }
  __syncthreads();

  // ---- Stage 1.5: L2-normalize q,k rows; fold clamped logit scale into q -
  for (int rr = 0; rr < 4; ++rr) {
    int rid = tid + rr * 512;                      // 2048 rows (Q then K)
    int h = (rid >> 6) & 15, n = rid & 63;
    _Float16* rowp = QKs + n * 1024 + (rid < 1024 ? 0 : 512) + h * 32;
    float ss = 0.0f;
#pragma unroll
    for (int d = 0; d < 32; ++d) { float v = (float)rowp[d]; ss += v * v; }
    float rs = 1.0f / fmaxf(sqrtf(ss), 1e-12f);
    if (rid < 1024) rs *= __expf(fminf(lscale[h], LN100));
#pragma unroll
    for (int d = 0; d < 32; ++d) rowp[d] = (_Float16)((float)rowp[d] * rs);
  }
  __syncthreads();

  // ---- Stage 2: per-head logits 64x64 via WMMA, kept in VGPRs -----------
  const int h = wave;
  v8f sacc[16];
#pragma unroll
  for (int mt = 0; mt < 4; ++mt) {
    v16h a = frag_a(QKs, 1024, mt * 16, h * 32, lane);          // q rows
#pragma unroll
    for (int nt = 0; nt < 4; ++nt) {
      v16h bb = frag_b(QKs + 512, 1024, nt * 16, h * 32, lane); // B = K^T
      v8f acc = {};
      sacc[mt * 4 + nt] = WMMA_F16(a, bb, acc);
    }
  }
  __syncthreads();   // all Q/K reads complete; Atn may overwrite X/QK-lo

  // add CPB bias + mask, write f16 logits [n][m][h] for head-axis softmax
  {
    const float* bh = bias16 + h * 4096;
    const float* mw = mask + (size_t)w * 4096;
#pragma unroll
    for (int mt = 0; mt < 4; ++mt)
#pragma unroll
      for (int nt = 0; nt < 4; ++nt) {
        int m = nt * 16 + (lane & 15);
#pragma unroll
        for (int r = 0; r < 8; ++r) {
          int n = mt * 16 + r + 8 * (lane >> 4);
          float s = sacc[mt * 4 + nt][r] + bh[n * 64 + m] + mw[n * 64 + m];
          Atn[(n * 64 + m) * 16 + h] = (_Float16)s;
        }
      }
  }
  __syncthreads();

  // ---- Stage 3: softmax over the 16 HEADS per (n,m); probs -> [h][n][m] --
  for (int i = 0; i < 8; ++i) {
    int p = tid + i * 512;                         // 4096 (n,m) pairs
    const _Float16* sp = Atn + p * 16;
    float vals[16], mx = -1e30f;
#pragma unroll
    for (int hh = 0; hh < 16; ++hh) { vals[hh] = (float)sp[hh]; mx = fmaxf(mx, vals[hh]); }
    float sum = 0.0f;
#pragma unroll
    for (int hh = 0; hh < 16; ++hh) { vals[hh] = __expf(vals[hh] - mx); sum += vals[hh]; }
    float inv = 1.0f / sum;
#pragma unroll
    for (int hh = 0; hh < 16; ++hh) Ps[hh * 4096 + p] = (_Float16)(vals[hh] * inv);
  }
  __syncthreads();

  // ---- Stage 4: out_h = P[h] (64x64) @ V[h] (64x32) via WMMA ------------
  {
    const _Float16* Ph = Ps + h * 4096;
    // Hoist the 4 B fragments (kt x dt), share across all mt.
    v16h bf[4];
#pragma unroll
    for (int kt = 0; kt < 2; ++kt)
#pragma unroll
      for (int dt = 0; dt < 2; ++dt)
        bf[kt * 2 + dt] = frag_b(VTs, 64, h * 32 + dt * 16, kt * 32, lane);
#pragma unroll
    for (int mt = 0; mt < 4; ++mt) {
      v8f acc[2] = {};
#pragma unroll
      for (int kt = 0; kt < 2; ++kt) {
        v16h a = frag_a(Ph, 64, mt * 16, kt * 32, lane);
#pragma unroll
        for (int dt = 0; dt < 2; ++dt)
          acc[dt] = WMMA_F16(a, bf[kt * 2 + dt], acc[dt]);
      }
#pragma unroll
      for (int dt = 0; dt < 2; ++dt) {
        int cc = h * 32 + dt * 16 + (lane & 15);
#pragma unroll
        for (int r = 0; r < 8; ++r) {
          int n = mt * 16 + r + 8 * (lane >> 4);
          AOs[n * 512 + cc] = (_Float16)acc[dt][r];  // AO region is dead Atn-lo
        }
      }
    }
  }
  __syncthreads();

  // ---- Stage 5: Y = AO (64x512) @ Wout (512x512), f32 stores -------------
  // Each wave owns one 16-row block (mt = wave>>2) and 8 col tiles; the A
  // fragment per k-step is shared across all 8 accumulators.
  {
    float* ob = out + (size_t)b * 64 * 512;
    const int mt  = wave >> 2;
    const int ct0 = (wave & 3) * 8;
    v8f acc[8] = {};
    for (int kt = 0; kt < 16; ++kt) {
      v16h a = frag_a(AOs, 512, mt * 16, kt * 32, lane);
#pragma unroll
      for (int t = 0; t < 8; ++t) {
        v16h bb = frag_b(woutT, 512, (ct0 + t) * 16, kt * 32, lane);
        acc[t] = WMMA_F16(a, bb, acc[t]);
      }
    }
#pragma unroll
    for (int t = 0; t < 8; ++t) {
      int c = (ct0 + t) * 16 + (lane & 15);
#pragma unroll
      for (int r = 0; r < 8; ++r) {
        int n = mt * 16 + r + 8 * (lane >> 4);
        ob[n * 512 + c] = acc[t][r];
      }
    }
  }
}

// ---------------------------------------------------------------------------
extern "C" void kernel_launch(void* const* d_in, const int* in_sizes, int n_in,
                              void* d_out, int out_size, void* d_ws, size_t ws_size,
                              hipStream_t stream) {
  (void)in_sizes; (void)n_in; (void)out_size; (void)ws_size;
  const float* x     = (const float*)d_in[0];
  const float* mask  = (const float*)d_in[1];
  const float* wqkv  = (const float*)d_in[2];
  const float* wout  = (const float*)d_in[3];
  const float* cpbw1 = (const float*)d_in[4];
  const float* cpbw2 = (const float*)d_in[5];
  const float* lsc   = (const float*)d_in[6];

  // Workspace: f16 transposed weights + CPB bias tables (~2.4 MB).
  _Float16* wqkvT = (_Float16*)d_ws;
  _Float16* woutT = wqkvT + 512 * 1536;
  float*    btab  = (float*)(woutT + 512 * 512);
  float*    b16   = btab + 225 * 16;

  prep_weights<<<512, 256, 0, stream>>>(wqkv, wout, wqkvT, woutT);
  cpb_table<<<225, 16, 0, stream>>>(cpbw1, cpbw2, btab);
  bias_sigmoid<<<256, 256, 0, stream>>>(btab, b16);

  const int shmem = 320 * 1024;   // full 320 KB WGP LDS per block
  hipFuncSetAttribute((const void*)fused_window_attn,
                      hipFuncAttributeMaxDynamicSharedMemorySize, shmem);
  fused_window_attn<<<2048, 512, shmem, stream>>>(x, mask, wqkvT, woutT, b16, lsc,
                                                  (float*)d_out);
}
